// ALaCarteClassifier_3934190043391
// MI455X (gfx1250) — matrix operand order repaired and checked by reading, not compiled, not run on hardware
//
#include <hip/hip_runtime.h>
#include <hip/hip_bf16.h>
#include <math.h>

// Problem constants (reference: VOCAB=200000, E=256, H=512, B=32, S=512, C=2)
#define B_  32
#define S_  512
#define E_  256
#define H_  512
#define G3_ 1536               // 3*H
#define M_  (B_ * S_)          // 16384 GEMM rows; row m = s*32 + b
#define NWG_ 32                // GRU workgroups (16 h-columns each)

typedef __attribute__((ext_vector_type(16))) __bf16        v16bf;
typedef __attribute__((ext_vector_type(8)))  float         v8f;
typedef __attribute__((ext_vector_type(4)))  unsigned int  v4u;

// 32-byte WMMA A/B fragment (16 bf16), filled via two 16B vector loads.
union Frag32 {
  v16bf bf;
  v4u   u4[2];
};

__device__ __forceinline__ unsigned short f2bf(float f) {
  unsigned u = __builtin_bit_cast(unsigned, f);
  u += 0x7fffu + ((u >> 16) & 1u);          // round-to-nearest-even
  return (unsigned short)(u >> 16);
}

__device__ __forceinline__ unsigned long long pack4bf(float4 f) {
  return (unsigned long long)f2bf(f.x)
       | ((unsigned long long)f2bf(f.y) << 16)
       | ((unsigned long long)f2bf(f.z) << 32)
       | ((unsigned long long)f2bf(f.w) << 48);
}

__device__ __forceinline__ float sigmoidf_(float x) {
  return 1.0f / (1.0f + __expf(-x));
}

// ---------------------------------------------------------------------------
// Kernel 1: induced = induction @ unk_vec   (256x256 * 256 -> 256)
// ---------------------------------------------------------------------------
__global__ void k_induced(const float* __restrict__ induction,
                          const float* __restrict__ unk,
                          float* __restrict__ induced) {
  __shared__ float u[E_];
  int t = threadIdx.x;
  u[t] = unk[t];
  __syncthreads();
  const float* row = induction + (size_t)t * E_;
  float s = 0.f;
  #pragma unroll 8
  for (int k = 0; k < E_; ++k) s = fmaf(row[k], u[k], s);
  induced[t] = s;
}

// ---------------------------------------------------------------------------
// Kernel 2: f32 -> bf16 weight conversion (4 elems/thread)
// ---------------------------------------------------------------------------
__global__ void k_cvt_bf16(const float* __restrict__ src,
                           unsigned short* __restrict__ dst, int n) {
  int i = (blockIdx.x * blockDim.x + threadIdx.x) << 2;
  if (i + 3 < n) {
    float4 f = *(const float4*)(src + i);
    *(unsigned long long*)(dst + i) = pack4bf(f);
  }
}

// ---------------------------------------------------------------------------
// Kernel 3: embedding gather + UNK replace, write bf16 A-matrix
// e_bf[m][k], m = s*32 + b
// ---------------------------------------------------------------------------
__global__ void k_embed(const int* __restrict__ x,
                        const float* __restrict__ emb,
                        const float* __restrict__ induced,
                        unsigned short* __restrict__ e) {
  int tid = blockIdx.x * blockDim.x + threadIdx.x;   // M_*64 threads total
  int m  = tid >> 6;
  int k0 = (tid & 63) << 2;
  int s = m >> 5, b = m & 31;
  int tok = x[b * S_ + s];
  const float* src = (tok < 0) ? (induced + k0)
                               : (emb + (size_t)tok * E_ + k0);
  float4 f = *(const float4*)src;
  *(unsigned long long*)(e + (size_t)m * E_ + k0) = pack4bf(f);
}

// ---------------------------------------------------------------------------
// Kernel 4: zero h(0) bf16 buffer and the grid-barrier counter
// ---------------------------------------------------------------------------
__global__ void k_zero(unsigned int* __restrict__ h0,
                       unsigned int* __restrict__ cnt) {
  int i = blockIdx.x * blockDim.x + threadIdx.x;
  if (i < (B_ * H_) / 2) h0[i] = 0u;          // 32KB of bf16 zeros
  if (i == 0) *cnt = 0u;
}

// ---------------------------------------------------------------------------
// Kernel 5: xi = e @ W_ih^T + b_ih, stored TRANSPOSED for the GRU:
//   xi_t[(s*G3 + col)*32 + b]   (b = batch = GEMM row % 32)
// 256 threads = 8 waves; block tile 128 rows x 64 cols; 4 C-tiles per wave.
// B fragments of each K-slice are preloaded into distinct registers so the
// scheduler can retire loads one-by-one behind earlier WMMAs (partial waits).
// ---------------------------------------------------------------------------
__global__ __launch_bounds__(256) void k_gemm_xi(
    const unsigned short* __restrict__ e,      // [M_, E_] bf16 row-major
    const unsigned short* __restrict__ wih,    // [G3_, E_] bf16 row-major
    const float* __restrict__ bih,             // [G3_]
    float* __restrict__ xi)                    // transposed layout above
{
  const int lane = threadIdx.x & 31;
  const int wv   = threadIdx.x >> 5;           // 0..7
  const int ln16 = lane & 15;
  const int hi   = lane >> 4;                  // lane group (0/1)
  const int m0   = blockIdx.x * 128 + wv * 16; // 16-row tile, same s for all
  const int n0   = blockIdx.y * 64;
  const int kbA  = hi * 8;
  const int kbB  = hi * 8;

  v8f zero = {};
  v8f acc[4];
  #pragma unroll
  for (int t = 0; t < 4; ++t) acc[t] = zero;

  const unsigned short* arow = e + (size_t)(m0 + ln16) * E_;
  #pragma unroll 2
  for (int k0 = 0; k0 < E_; k0 += 32) {
    Frag32 a;
    a.u4[0] = *(const v4u*)(arow + k0 + kbA);
    a.u4[1] = *(const v4u*)(arow + k0 + kbA + 16);
    Frag32 bb[4];
    #pragma unroll
    for (int t = 0; t < 4; ++t) {
      const unsigned short* brow =
          wih + (size_t)(n0 + t * 16 + ln16) * E_ + k0 + kbB;
      bb[t].u4[0] = *(const v4u*)(brow);
      bb[t].u4[1] = *(const v4u*)(brow + 16);
    }
    #pragma unroll
    for (int t = 0; t < 4; ++t)
      acc[t] = __builtin_amdgcn_wmma_f32_16x16x32_bf16(
                   false, a.bf, false, bb[t].bf, (short)0, acc[t], false, false);
  }

  const int s_idx = m0 >> 5;                   // sequence position
  const int b0    = (m0 & 31) + hi * 8;        // batch base of VGPR v=0
  #pragma unroll
  for (int t = 0; t < 4; ++t) {
    int col = n0 + t * 16 + ln16;
    float bias = bih[col];
    float* dst = xi + ((size_t)s_idx * G3_ + col) * B_ + b0;
    float4 o0, o1;
    o0.x = acc[t][0] + bias; o0.y = acc[t][1] + bias;
    o0.z = acc[t][2] + bias; o0.w = acc[t][3] + bias;
    o1.x = acc[t][4] + bias; o1.y = acc[t][5] + bias;
    o1.z = acc[t][6] + bias; o1.w = acc[t][7] + bias;
    *(float4*)(dst)     = o0;
    *(float4*)(dst + 4) = o1;
  }
}

// ---------------------------------------------------------------------------
// Device-wide barrier for the GRU (32 co-resident blocks).
// ---------------------------------------------------------------------------
__device__ __forceinline__ void grid_barrier(unsigned int* cnt, unsigned int target) {
  __threadfence();                              // own stores -> device scope
  __syncthreads();
  if (threadIdx.x == 0) {
    __hip_atomic_fetch_add(cnt, 1u, __ATOMIC_RELEASE, __HIP_MEMORY_SCOPE_AGENT);
    while (__hip_atomic_load(cnt, __ATOMIC_ACQUIRE, __HIP_MEMORY_SCOPE_AGENT) < target) {
      __builtin_amdgcn_s_sleep(1);
    }
  }
  __syncthreads();
  (void)__hip_atomic_load(cnt, __ATOMIC_ACQUIRE, __HIP_MEMORY_SCOPE_AGENT);
}

// ---------------------------------------------------------------------------
// Kernel 6: GRU recurrence, 32 blocks x 64 threads (2 waves).
// Block owns 16 h-columns; its W_hh slice (3 gates x 16 rows x 512) = 48KB
// lives in LDS (loaded once). h is double-buffered bf16 in global memory;
// steps are separated by grid_barrier. Wave wv handles M-tile wv (rows
// 16wv..16wv+15): 3 C-tiles (r,z,n), 48 WMMAs per step.
// ---------------------------------------------------------------------------
__global__ __launch_bounds__(64) void k_gru(
    const unsigned short* __restrict__ whh,    // [G3_, H_] bf16 row-major
    const float* __restrict__ bhh,             // [G3_]
    const float* __restrict__ xi,              // transposed xi (see k_gemm_xi)
    unsigned short* __restrict__ hg,           // [2][B_*H_] bf16, hg[0] zeroed
    unsigned int* __restrict__ cnt,            // barrier counter (zeroed)
    float* __restrict__ pooled)                // [B_, H_] f32
{
  __shared__ unsigned short Wlds[48 * H_];     // 48KB: rows g*16+c, c=0..15

  const int lane = threadIdx.x & 31;
  const int wv   = threadIdx.x >> 5;           // 0/1 == M-tile
  const int ln16 = lane & 15;
  const int hi   = lane >> 4;
  const int c0   = blockIdx.x * 16;
  const int col  = c0 + ln16;                  // this lane's h column
  const int kbA  = hi * 8;
  const int kbB  = hi * 8;

  // stage W_hh slice into LDS: local row r = g*16 + c  <->  global g*512+c0+c
  for (int r = 0; r < 48; ++r) {
    int g = r >> 4, cc = r & 15;
    const unsigned short* src = whh + (size_t)(g * H_ + c0 + cc) * H_;
    *(v4u*)&Wlds[r * H_ + threadIdx.x * 8] = *(const v4u*)(src + threadIdx.x * 8);
  }
  __syncthreads();

  const float bhr = bhh[col];
  const float bhz = bhh[H_ + col];
  const float bhn = bhh[2 * H_ + col];

  float h[8], hmax[8];
  #pragma unroll
  for (int i = 0; i < 8; ++i) { h[i] = 0.f; hmax[i] = -INFINITY; }

  v8f zero = {};
  for (int t = 0; t < S_; ++t) {
    const unsigned short* hr = hg + (size_t)(t & 1) * (B_ * H_);
    unsigned short*       hw = hg + (size_t)((t & 1) ^ 1) * (B_ * H_);

    // prefetch next step's xi lines for this lane (global_prefetch_b8)
    if (t + 1 < S_) {
      #pragma unroll
      for (int g = 0; g < 3; ++g)
        __builtin_prefetch(
            xi + ((size_t)(t + 1) * G3_ + g * H_ + col) * B_ + wv * 16 + hi * 8,
            0, 1);
    }

    v8f acc[3];
    #pragma unroll
    for (int g = 0; g < 3; ++g) acc[g] = zero;

    #pragma unroll 2
    for (int k0 = 0; k0 < H_; k0 += 32) {
      Frag32 a;
      const unsigned short* ap =
          hr + (size_t)(wv * 16 + ln16) * H_ + k0 + kbA;
      a.u4[0] = *(const v4u*)(ap);
      a.u4[1] = *(const v4u*)(ap + 16);
      Frag32 bb[3];
      #pragma unroll
      for (int g = 0; g < 3; ++g) {
        const unsigned short* brow = &Wlds[(g * 16 + ln16) * H_ + k0 + kbB];
        bb[g].u4[0] = *(const v4u*)(brow);
        bb[g].u4[1] = *(const v4u*)(brow + 16);
      }
      #pragma unroll
      for (int g = 0; g < 3; ++g)
        acc[g] = __builtin_amdgcn_wmma_f32_16x16x32_bf16(
                     false, a.bf, false, bb[g].bf, (short)0, acc[g], false, false);
    }

    // gates (torch order r,z,n); xi already contains b_ih
    float xv[3][8];
    #pragma unroll
    for (int g = 0; g < 3; ++g) {
      const float* xg =
          xi + ((size_t)t * G3_ + g * H_ + col) * B_ + wv * 16 + hi * 8;
      float4 q0 = *(const float4*)(xg);
      float4 q1 = *(const float4*)(xg + 4);
      xv[g][0] = q0.x; xv[g][1] = q0.y; xv[g][2] = q0.z; xv[g][3] = q0.w;
      xv[g][4] = q1.x; xv[g][5] = q1.y; xv[g][6] = q1.z; xv[g][7] = q1.w;
    }
    #pragma unroll
    for (int v = 0; v < 8; ++v) {
      int m = wv * 16 + hi * 8 + v;            // batch row (C layout: v -> M)
      float gr = sigmoidf_(xv[0][v] + acc[0][v] + bhr);
      float gz = sigmoidf_(xv[1][v] + acc[1][v] + bhz);
      float gn = tanhf(xv[2][v] + gr * (acc[2][v] + bhn));
      float hn = (1.f - gz) * gn + gz * h[v];
      h[v] = hn;
      hmax[v] = fmaxf(hmax[v], hn);
      hw[m * H_ + col] = f2bf(hn);
    }

    grid_barrier(cnt, (unsigned)NWG_ * (unsigned)(t + 1));
  }

  #pragma unroll
  for (int v = 0; v < 8; ++v)
    pooled[(wv * 16 + hi * 8 + v) * H_ + col] = hmax[v];
}

// ---------------------------------------------------------------------------
// Kernel 7: out = pooled @ W_proj^T + b_proj   [32,512]x[512,2]
// ---------------------------------------------------------------------------
__global__ void k_proj(const float* __restrict__ pooled,
                       const float* __restrict__ wproj,
                       const float* __restrict__ bproj,
                       float* __restrict__ out) {
  int t = threadIdx.x;            // 64 threads: t = b*2 + c
  int b = t >> 1, c = t & 1;
  const float* p = pooled + b * H_;
  const float* w = wproj + c * H_;
  float s = bproj[c];
  #pragma unroll 8
  for (int k = 0; k < H_; ++k) s = fmaf(p[k], w[k], s);
  out[t] = s;
}

// ---------------------------------------------------------------------------
extern "C" void kernel_launch(void* const* d_in, const int* in_sizes, int n_in,
                              void* d_out, int out_size, void* d_ws, size_t ws_size,
                              hipStream_t stream) {
  const int*   x     = (const int*)d_in[0];
  const float* emb   = (const float*)d_in[1];
  const float* unk   = (const float*)d_in[2];
  const float* indm  = (const float*)d_in[3];
  const float* wih   = (const float*)d_in[4];
  const float* whh   = (const float*)d_in[5];
  const float* bih   = (const float*)d_in[6];
  const float* bhh   = (const float*)d_in[7];
  const float* wproj = (const float*)d_in[8];
  const float* bproj = (const float*)d_in[9];
  float* out = (float*)d_out;

  // workspace layout (total need ~117 MB; xi dominates and fits in L2)
  char* ws = (char*)d_ws;
  float*          induced = (float*)(ws + 0);                   //   1 KB
  float*          pooled  = (float*)(ws + 4096);                //  64 KB
  unsigned short* wih_bf  = (unsigned short*)(ws + 0x20000);    // 768 KB
  unsigned short* whh_bf  = (unsigned short*)(ws + 0x100000);   // 1.5 MB
  unsigned short* h_glob  = (unsigned short*)(ws + 0x300000);   //  64 KB (x2 buf)
  unsigned int*   cnt     = (unsigned int*)(ws + 0x310000);     //   4 B
  unsigned short* e_bf    = (unsigned short*)(ws + 0x400000);   //   8 MB
  float*          xi      = (float*)(ws + 0x1000000);           // 100.7 MB

  k_induced<<<1, 256, 0, stream>>>(indm, unk, induced);
  k_cvt_bf16<<<(G3_ * E_) / 1024, 256, 0, stream>>>(wih, wih_bf, G3_ * E_);
  k_cvt_bf16<<<(G3_ * H_) / 1024, 256, 0, stream>>>(whh, whh_bf, G3_ * H_);
  k_embed<<<(M_ * 64) / 256, 256, 0, stream>>>(x, emb, induced, e_bf);
  k_zero<<<32, 256, 0, stream>>>((unsigned int*)h_glob, cnt);

  dim3 g(M_ / 128, G3_ / 64);
  k_gemm_xi<<<g, 256, 0, stream>>>(e_bf, wih_bf, bih, xi);

  k_gru<<<NWG_, 64, 0, stream>>>(whh_bf, bhh, xi, h_glob, cnt, pooled);
  k_proj<<<1, 64, 0, stream>>>(pooled, wproj, bproj, out);
}
